// SwinTransformerBlock_25640954757307
// MI455X (gfx1250) — matrix operand order repaired and checked by reading, not compiled, not run on hardware
//
#include <hip/hip_runtime.h>
#include <hip/hip_bf16.h>
#include <math.h>

typedef __bf16 bf16;
typedef __attribute__((ext_vector_type(16))) __bf16 v16bf;
typedef __attribute__((ext_vector_type(8)))  __bf16 v8bf;
typedef __attribute__((ext_vector_type(8)))  float  v8f;

#define WMMA_BF16(a, b, c) \
  __builtin_amdgcn_wmma_f32_16x16x32_bf16(false, (a), false, (b), (short)0, (c), false, false)

// ---------------------------------------------------------------------------
// Fragment loaders (wave32, 16x16x32 bf16 WMMA)
// A (16x32, 16-bit): lane<16 -> M=lane,    K = {k0+0..7, k0+16..23}
//                    lane>=16 -> M=lane-16, K = {k0+8..15, k0+24..31}
// B (32x16, 16-bit): lane<16 -> N=lane,    K = {k0+0..7, k0+16..23}
//                    lane>=16 -> N=lane-16, K = {k0+8..15, k0+24..31}
// Both implemented as two 16-byte loads per lane.
// ---------------------------------------------------------------------------
__device__ __forceinline__ v16bf frag_a(const bf16* base, int ld, int r0, int k0) {
  const int lane = threadIdx.x & 31;
  const bf16* p = base + (size_t)(r0 + (lane & 15)) * ld + (k0 + ((lane >> 4) << 3));
  union { v16bf v; v8bf h[2]; } u;
  u.h[0] = *(const v8bf*)(p);
  u.h[1] = *(const v8bf*)(p + 16);
  return u.v;
}

__device__ __forceinline__ v16bf frag_b(const bf16* base, int ld, int n0, int k0) {
  const int lane = threadIdx.x & 31;
  const bf16* p = base + (size_t)(n0 + (lane & 15)) * ld + (k0 + ((lane >> 4) << 3));
  union { v16bf v; v8bf h[2]; } u;
  u.h[0] = *(const v8bf*)(p);
  u.h[1] = *(const v8bf*)(p + 16);
  return u.v;
}

// ---------------------------------------------------------------------------
// fp32 -> bf16 weight conversion
// ---------------------------------------------------------------------------
__global__ void cvt_kernel(const float* __restrict__ in, bf16* __restrict__ out, int n) {
  int i = blockIdx.x * 256 + threadIdx.x;
  if (i < n) out[i] = (bf16)in[i];
}

// ---------------------------------------------------------------------------
// LayerNorm1 + window partition.  x:(B,C,H,W) fp32 -> xln:(B_*64, 192) bf16
// Block handles 32 consecutive w positions at fixed (b,h); coalesced channel
// loads staged through LDS, then wave32 shuffle-reduce LN per token.
// ---------------------------------------------------------------------------
__global__ void ln1_part_kernel(const float* __restrict__ x, const float* __restrict__ gw,
                                const float* __restrict__ gb, bf16* __restrict__ xln) {
  const int b = blockIdx.z, h = blockIdx.y, w0 = blockIdx.x * 32;
  __shared__ float tile[32][196];
  const int t = threadIdx.x;
  const int wi = t & 31, crow = t >> 5;
  for (int c0 = 0; c0 < 192; c0 += 8) {
    int c = c0 + crow;
    tile[wi][c] = x[(((size_t)b * 192 + c) * 256 + h) * 256 + (w0 + wi)];
  }
  __syncthreads();
  const int wid = t >> 5, lane = t & 31;
  for (int s = 0; s < 4; ++s) {
    const int wi2 = wid * 4 + s;
    float v[6];
    float sum = 0.f;
#pragma unroll
    for (int e = 0; e < 6; ++e) { v[e] = tile[wi2][lane * 6 + e]; sum += v[e]; }
#pragma unroll
    for (int off = 16; off >= 1; off >>= 1) sum += __shfl_xor(sum, off, 32);
    const float mu = sum * (1.f / 192.f);
    float var = 0.f;
#pragma unroll
    for (int e = 0; e < 6; ++e) { float d = v[e] - mu; var += d * d; }
#pragma unroll
    for (int off = 16; off >= 1; off >>= 1) var += __shfl_xor(var, off, 32);
    const float rs = rsqrtf(var * (1.f / 192.f) + 1e-5f);
    const int ww = w0 + wi2;
    const int win = ((b * 32) + (h >> 3)) * 32 + (ww >> 3);
    const size_t row = (size_t)win * 64 + (h & 7) * 8 + (ww & 7);
    bf16* dst = xln + row * 192 + lane * 6;
#pragma unroll
    for (int e = 0; e < 6; ++e) {
      int c = lane * 6 + e;
      dst[e] = (bf16)((v[e] - mu) * rs * gw[c] + gb[c]);
    }
  }
}

// ---------------------------------------------------------------------------
// QKV GEMM: (262144 x 192) @ (192 x 576)^T-weights.  One wave = 16x64 tile.
// q scaled by hd^-0.5; v stored transposed per head (32 x 64).
// ---------------------------------------------------------------------------
__global__ void qkv_gemm_kernel(const bf16* __restrict__ A, const bf16* __restrict__ W,
                                const float* __restrict__ bias,
                                bf16* __restrict__ qo, bf16* __restrict__ ko,
                                bf16* __restrict__ vT) {
  const int lane = threadIdx.x & 31, wid = threadIdx.x >> 5;
  const int m0 = blockIdx.x * 128 + wid * 16;
  const int n0 = blockIdx.y * 64;
  const v8f zero = {0.f, 0.f, 0.f, 0.f, 0.f, 0.f, 0.f, 0.f};
  v8f acc[4] = {zero, zero, zero, zero};
  for (int k0 = 0; k0 < 192; k0 += 32) {
    v16bf a = frag_a(A, 192, m0, k0);
#pragma unroll
    for (int j = 0; j < 4; ++j) {
      v16bf bf_ = frag_b(W, 192, n0 + j * 16, k0);
      acc[j] = WMMA_BF16(a, bf_, acc[j]);
    }
  }
  const float scale = 0.17677669529663687f;  // 32^-0.5
#pragma unroll
  for (int j = 0; j < 4; ++j) {
    const int col = n0 + j * 16 + (lane & 15);
    const int which = col / 192;
    const int c = col - which * 192;
    const int hh = c >> 5, d = c & 31;
    const float bv = bias[col];
#pragma unroll
    for (int i = 0; i < 8; ++i) {
      const int row = m0 + ((lane >> 4) << 3) + i;
      const int wn = row >> 6, tok = row & 63;
      const float v = acc[j][i] + bv;
      const size_t base = (size_t)wn * 6 + hh;
      if (which == 0)      qo[(base * 64 + tok) * 32 + d] = (bf16)(v * scale);
      else if (which == 1) ko[(base * 64 + tok) * 32 + d] = (bf16)v;
      else                 vT[(base * 32 + d) * 64 + tok] = (bf16)v;
    }
  }
}

// ---------------------------------------------------------------------------
// Attention per (window, head): S = q k^T (WMMA K=32), fp32 softmax via
// 16-lane butterflies, P restaged through wave-private LDS, out = P V (WMMA).
// ---------------------------------------------------------------------------
__global__ void attn_kernel(const bf16* __restrict__ q, const bf16* __restrict__ k,
                            const bf16* __restrict__ vT, bf16* __restrict__ ao) {
  const int whid = blockIdx.x;           // window*6 + head
  const int head = whid % 6;
  const int win  = whid / 6;
  const bf16* qb = q  + (size_t)whid * 64 * 32;
  const bf16* kb = k  + (size_t)whid * 64 * 32;
  const bf16* vb = vT + (size_t)whid * 32 * 64;
  __shared__ __align__(16) bf16 plds[4][16][72];
  const int wid = threadIdx.x >> 5, lane = threadIdx.x & 31;
  const int m0 = wid * 16;
  const v8f zero = {0.f, 0.f, 0.f, 0.f, 0.f, 0.f, 0.f, 0.f};

  v8f s[4] = {zero, zero, zero, zero};
  v16bf aq = frag_a(qb, 32, m0, 0);
#pragma unroll
  for (int j = 0; j < 4; ++j) {
    v16bf bk = frag_b(kb, 32, j * 16, 0);   // B column n = row n of k
    s[j] = WMMA_BF16(aq, bk, s[j]);
  }
  const int n = lane & 15;
  const int rbase = (lane >> 4) << 3;
#pragma unroll
  for (int i = 0; i < 8; ++i) {
    float mx = fmaxf(fmaxf(s[0][i], s[1][i]), fmaxf(s[2][i], s[3][i]));
#pragma unroll
    for (int off = 8; off >= 1; off >>= 1) mx = fmaxf(mx, __shfl_xor(mx, off, 16));
    float e0 = __expf(s[0][i] - mx), e1 = __expf(s[1][i] - mx);
    float e2 = __expf(s[2][i] - mx), e3 = __expf(s[3][i] - mx);
    float sum = e0 + e1 + e2 + e3;
#pragma unroll
    for (int off = 8; off >= 1; off >>= 1) sum += __shfl_xor(sum, off, 16);
    const float inv = 1.f / sum;
    const int r = rbase + i;
    plds[wid][r][0 * 16 + n] = (bf16)(e0 * inv);
    plds[wid][r][1 * 16 + n] = (bf16)(e1 * inv);
    plds[wid][r][2 * 16 + n] = (bf16)(e2 * inv);
    plds[wid][r][3 * 16 + n] = (bf16)(e3 * inv);
  }
  // wave-private LDS: in-order DS ops, no cross-wave barrier required
  v8f o[2] = {zero, zero};
  const bf16* pl = &plds[wid][0][0];
#pragma unroll
  for (int kk = 0; kk < 64; kk += 32) {
    v16bf pa = frag_a(pl, 72, 0, kk);
#pragma unroll
    for (int nt = 0; nt < 2; ++nt) {
      v16bf bv = frag_b(vb, 64, nt * 16, kk);  // vT row d = column d of v
      o[nt] = WMMA_BF16(pa, bv, o[nt]);
    }
  }
#pragma unroll
  for (int nt = 0; nt < 2; ++nt) {
    const int ch = head * 32 + nt * 16 + n;
#pragma unroll
    for (int i = 0; i < 8; ++i) {
      const size_t row = (size_t)win * 64 + m0 + rbase + i;
      ao[row * 192 + ch] = (bf16)o[nt][i];
    }
  }
}

// ---------------------------------------------------------------------------
// Proj GEMM + window reverse + residual -> x1 channels-last fp32
// ---------------------------------------------------------------------------
__global__ void proj_gemm_kernel(const bf16* __restrict__ A, const bf16* __restrict__ W,
                                 const float* __restrict__ bias, const float* __restrict__ xin,
                                 float* __restrict__ x1) {
  const int lane = threadIdx.x & 31, wid = threadIdx.x >> 5;
  const int m0 = blockIdx.x * 128 + wid * 16;
  const int n0 = blockIdx.y * 64;
  const v8f zero = {0.f, 0.f, 0.f, 0.f, 0.f, 0.f, 0.f, 0.f};
  v8f acc[4] = {zero, zero, zero, zero};
  for (int k0 = 0; k0 < 192; k0 += 32) {
    v16bf a = frag_a(A, 192, m0, k0);
#pragma unroll
    for (int j = 0; j < 4; ++j) {
      v16bf bf_ = frag_b(W, 192, n0 + j * 16, k0);
      acc[j] = WMMA_BF16(a, bf_, acc[j]);
    }
  }
#pragma unroll
  for (int j = 0; j < 4; ++j) {
    const int c = n0 + j * 16 + (lane & 15);
    const float bv = bias[c];
#pragma unroll
    for (int i = 0; i < 8; ++i) {
      const int row = m0 + ((lane >> 4) << 3) + i;
      const int wn = row >> 6, tok = row & 63;
      const int b = wn >> 10, rem = wn & 1023;
      const int hh = ((rem >> 5) << 3) + (tok >> 3);
      const int ww = ((rem & 31) << 3) + (tok & 7);
      const float res = xin[(((size_t)b * 192 + c) * 256 + hh) * 256 + ww];
      x1[(((size_t)b * 256 + hh) * 256 + ww) * 192 + c] = acc[j][i] + bv + res;
    }
  }
}

// ---------------------------------------------------------------------------
// LayerNorm2: x1 channels-last fp32 -> bf16 (wave per token)
// ---------------------------------------------------------------------------
__global__ void ln2_kernel(const float* __restrict__ x1, const float* __restrict__ gw,
                           const float* __restrict__ gb, bf16* __restrict__ xln) {
  const int wid = threadIdx.x >> 5, lane = threadIdx.x & 31;
  const size_t row = (size_t)blockIdx.x * 8 + wid;
  const float* p = x1 + row * 192 + lane * 6;
  float v[6];
  float sum = 0.f;
#pragma unroll
  for (int e = 0; e < 6; ++e) { v[e] = p[e]; sum += v[e]; }
#pragma unroll
  for (int off = 16; off >= 1; off >>= 1) sum += __shfl_xor(sum, off, 32);
  const float mu = sum * (1.f / 192.f);
  float var = 0.f;
#pragma unroll
  for (int e = 0; e < 6; ++e) { float d = v[e] - mu; var += d * d; }
#pragma unroll
  for (int off = 16; off >= 1; off >>= 1) var += __shfl_xor(var, off, 32);
  const float rs = rsqrtf(var * (1.f / 192.f) + 1e-5f);
  bf16* dst = xln + row * 192 + lane * 6;
#pragma unroll
  for (int e = 0; e < 6; ++e) {
    int c = lane * 6 + e;
    dst[e] = (bf16)((v[e] - mu) * rs * gw[c] + gb[c]);
  }
}

// ---------------------------------------------------------------------------
// MLP fc1 + exact GELU -> h1 bf16 (262144 x 768)
// ---------------------------------------------------------------------------
__global__ void mlp1_gemm_kernel(const bf16* __restrict__ A, const bf16* __restrict__ W,
                                 const float* __restrict__ bias, bf16* __restrict__ h1) {
  const int lane = threadIdx.x & 31, wid = threadIdx.x >> 5;
  const int m0 = blockIdx.x * 128 + wid * 16;
  const int n0 = blockIdx.y * 64;
  const v8f zero = {0.f, 0.f, 0.f, 0.f, 0.f, 0.f, 0.f, 0.f};
  v8f acc[4] = {zero, zero, zero, zero};
  for (int k0 = 0; k0 < 192; k0 += 32) {
    v16bf a = frag_a(A, 192, m0, k0);
#pragma unroll
    for (int j = 0; j < 4; ++j) {
      v16bf bf_ = frag_b(W, 192, n0 + j * 16, k0);
      acc[j] = WMMA_BF16(a, bf_, acc[j]);
    }
  }
#pragma unroll
  for (int j = 0; j < 4; ++j) {
    const int col = n0 + j * 16 + (lane & 15);
    const float bv = bias[col];
#pragma unroll
    for (int i = 0; i < 8; ++i) {
      const size_t row = (size_t)m0 + ((lane >> 4) << 3) + i;
      const float v = acc[j][i] + bv;
      const float g = 0.5f * v * (1.f + erff(v * 0.70710678118654752f));
      h1[row * 768 + col] = (bf16)g;
    }
  }
}

// ---------------------------------------------------------------------------
// MLP fc2 + residual -> out (B,C,H,W) fp32
// ---------------------------------------------------------------------------
__global__ void mlp2_gemm_kernel(const bf16* __restrict__ A, const bf16* __restrict__ W,
                                 const float* __restrict__ bias, const float* __restrict__ x1,
                                 float* __restrict__ out) {
  const int lane = threadIdx.x & 31, wid = threadIdx.x >> 5;
  const int m0 = blockIdx.x * 128 + wid * 16;
  const int n0 = blockIdx.y * 64;
  const v8f zero = {0.f, 0.f, 0.f, 0.f, 0.f, 0.f, 0.f, 0.f};
  v8f acc[4] = {zero, zero, zero, zero};
  for (int k0 = 0; k0 < 768; k0 += 32) {
    v16bf a = frag_a(A, 768, m0, k0);
#pragma unroll
    for (int j = 0; j < 4; ++j) {
      v16bf bf_ = frag_b(W, 768, n0 + j * 16, k0);
      acc[j] = WMMA_BF16(a, bf_, acc[j]);
    }
  }
#pragma unroll
  for (int j = 0; j < 4; ++j) {
    const int c = n0 + j * 16 + (lane & 15);
    const float bv = bias[c];
#pragma unroll
    for (int i = 0; i < 8; ++i) {
      const int row = m0 + ((lane >> 4) << 3) + i;
      const int b = row >> 16;
      const int hw = row & 65535;
      const int hh = hw >> 8, ww = hw & 255;
      const float v = acc[j][i] + bv + x1[(size_t)row * 192 + c];
      out[(((size_t)b * 192 + c) * 256 + hh) * 256 + ww] = v;
    }
  }
}

// ---------------------------------------------------------------------------
extern "C" void kernel_launch(void* const* d_in, const int* in_sizes, int n_in,
                              void* d_out, int out_size, void* d_ws, size_t ws_size,
                              hipStream_t stream) {
  (void)in_sizes; (void)n_in; (void)out_size; (void)ws_size;
  const float* x      = (const float*)d_in[0];
  const float* n1w    = (const float*)d_in[1];
  const float* n1b    = (const float*)d_in[2];
  const float* qkv_w  = (const float*)d_in[3];
  const float* qkv_b  = (const float*)d_in[4];
  const float* proj_w = (const float*)d_in[5];
  const float* proj_b = (const float*)d_in[6];
  const float* n2w    = (const float*)d_in[7];
  const float* n2b    = (const float*)d_in[8];
  const float* w1     = (const float*)d_in[9];
  const float* b1     = (const float*)d_in[10];
  const float* w2     = (const float*)d_in[11];
  const float* b2     = (const float*)d_in[12];
  float* out = (float*)d_out;

  char* ws = (char*)d_ws;
  size_t off = 0;
  auto alloc = [&](size_t bytes) -> void* {
    void* p = ws + off;
    off += (bytes + 255) & ~(size_t)255;
    return p;
  };
  const size_t TOK = 262144;  // 4*256*256 tokens
  bf16*  xln   = (bf16*)alloc(TOK * 192 * 2);   // LN output (reused for LN2)
  bf16*  qb    = (bf16*)alloc(TOK * 192 * 2);   // q  (B_,NH,64,32)
  bf16*  kb    = (bf16*)alloc(TOK * 192 * 2);   // k  (B_,NH,64,32)
  bf16*  vTb   = (bf16*)alloc(TOK * 192 * 2);   // vT (B_,NH,32,64)
  bf16*  attn  = (bf16*)alloc(TOK * 192 * 2);   // attention out (rows, 192)
  float* x1    = (float*)alloc(TOK * 192 * 4);  // channels-last residual
  bf16*  wqkv  = (bf16*)alloc(576 * 192 * 2);
  bf16*  wproj = (bf16*)alloc(192 * 192 * 2);
  bf16*  wm1   = (bf16*)alloc(768 * 192 * 2);
  bf16*  wm2   = (bf16*)alloc(192 * 768 * 2);
  bf16*  h1    = qb;  // overlay: q..attn (4*100.7MB) == 262144*768 bf16 exactly

  cvt_kernel<<<dim3((110592 + 255) / 256), 256, 0, stream>>>(qkv_w, wqkv, 110592);
  cvt_kernel<<<dim3((36864 + 255) / 256), 256, 0, stream>>>(proj_w, wproj, 36864);
  cvt_kernel<<<dim3((147456 + 255) / 256), 256, 0, stream>>>(w1, wm1, 147456);
  cvt_kernel<<<dim3((147456 + 255) / 256), 256, 0, stream>>>(w2, wm2, 147456);

  ln1_part_kernel<<<dim3(8, 256, 4), 256, 0, stream>>>(x, n1w, n1b, xln);
  qkv_gemm_kernel<<<dim3(2048, 9), 256, 0, stream>>>(xln, wqkv, qkv_b, qb, kb, vTb);
  attn_kernel<<<dim3(24576), 128, 0, stream>>>(qb, kb, vTb, attn);
  proj_gemm_kernel<<<dim3(2048, 3), 256, 0, stream>>>(attn, wproj, proj_b, x, x1);
  ln2_kernel<<<dim3(32768), 256, 0, stream>>>(x1, n2w, n2b, xln);
  mlp1_gemm_kernel<<<dim3(2048, 12), 256, 0, stream>>>(xln, wm1, b1, h1);
  mlp2_gemm_kernel<<<dim3(2048, 3), 256, 0, stream>>>(h1, wm2, b2, x1, out);
}
